// LSTMAuto_22016002359489
// MI455X (gfx1250) — matrix operand order repaired and checked by reading, not compiled, not run on hardware
//
#include <hip/hip_runtime.h>
#include <hip/hip_bf16.h>
#include <stddef.h>

// ---------------------------------------------------------------------------
// LSTM autoregressive sampler for MI455X (gfx1250, wave32, WMMA).
// T=128 steps, IN=64, H=2048, OUT=4096, L=2, batch=1.
//
// Weights converted once to bf16 (119 MB -> resident in the 192 MB L2); each
// step streams them through v_wmma_f32_16x16x32_bf16 matvecs (B = the vector
// replicated across all 16 columns).  ~1 FLOP/byte -> purely memory bound, so
// the wasted WMMA columns are free and each wmma consumes a 1 KB weight tile.
// K dimensions are compile-time template parameters so the inner loops are
// clean unrolled load+wmma clauses with no remainder scaffolding, and two
// ping-pong accumulators break the WMMA C->C dependency chain.
// ---------------------------------------------------------------------------

typedef __attribute__((ext_vector_type(16))) __bf16 v16bf;
typedef __attribute__((ext_vector_type(8)))  float  v8f;

#define TSTEPS 128
#define INSZ   64
#define HID    2048
#define GATES  8192   // 4*HID
#define VOCAB  4096
#define K0PAD  96     // IN+1=65 padded up to multiple of 32

__device__ __forceinline__ __bf16 f2bf(float f) {
    union { float f; unsigned int u; } x; x.f = f;
    unsigned int r = x.u + 0x7FFFu + ((x.u >> 16) & 1u);   // round-nearest-even
    unsigned short h = (unsigned short)(r >> 16);
    __bf16 out;
    __builtin_memcpy(&out, &h, 2);
    return out;
}

// ------------------------- weight preparation ------------------------------

__global__ void cvt_bf16_kernel(const float* __restrict__ src,
                                __bf16* __restrict__ dst, int n) {
    for (int i = blockIdx.x * blockDim.x + threadIdx.x; i < n;
         i += gridDim.x * blockDim.x)
        dst[i] = f2bf(src[i]);
}

// Wih0 [8192 x 65] -> bf16 [8192 x 96], zero padded K
__global__ void cvt_pad_wih0_kernel(const float* __restrict__ src,
                                    __bf16* __restrict__ dst) {
    int n = GATES * K0PAD;
    for (int i = blockIdx.x * blockDim.x + threadIdx.x; i < n;
         i += gridDim.x * blockDim.x) {
        int r = i / K0PAD, k = i - r * K0PAD;
        float v = (k < (INSZ + 1)) ? src[r * (INSZ + 1) + k] : 0.0f;
        dst[i] = f2bf(v);
    }
}

__global__ void bias_sum_kernel(const float* __restrict__ a,
                                const float* __restrict__ b,
                                float* __restrict__ out, int n) {
    for (int i = blockIdx.x * blockDim.x + threadIdx.x; i < n;
         i += gridDim.x * blockDim.x)
        out[i] = a[i] + b[i];
}

// ------------------------------ state init ---------------------------------

__global__ void init_state_kernel(const float* __restrict__ h0in,
                                  const float* __restrict__ c0in,
                                  const float* __restrict__ input,
                                  float* h0f, float* c0f, float* h1f, float* c1f,
                                  __bf16* h0b, __bf16* h1b,
                                  __bf16* xa, float* prev) {
    int i = blockIdx.x * blockDim.x + threadIdx.x;
    if (i < HID) {
        float a = h0in[i], b = h0in[HID + i];
        h0f[i] = a;            h1f[i] = b;
        c0f[i] = c0in[i];      c1f[i] = c0in[HID + i];
        h0b[i] = f2bf(a);      h1b[i] = f2bf(b);
    }
    if (i == 0) prev[0] = 1.0f;
    if (i < K0PAD) {   // xa for t=0: [prev=1.0, input[0,:], zeros]
        float v = (i == 0) ? 1.0f : (i <= INSZ ? input[i - 1] : 0.0f);
        xa[i] = f2bf(v);
    }
}

// --------------------------- WMMA dual matvec ------------------------------
// y[rows] = W1[rows x K1] * x1 + W2[rows x K2] * x2 + bias.
// One wave per 16 rows.  A-tile per the 16-bit A layout (lanes 0-15 hold K
// {0..7,16..23}; lanes 16-31 hold K {8..15,24..31}); B = x replicated across
// all 16 columns (same K order as A), so all
// columns of D carry the same dot product.

template <int ACC>
__device__ __forceinline__ void gemv_loop(const __bf16* __restrict__ W,
                                          const __bf16* __restrict__ x,
                                          int K,           // compile-time const
                                          int row_m, int hi,
                                          v8f acc[2]) {
    const __bf16* wrow = W + (size_t)row_m * K + hi * 8;
    const __bf16* xv   = x + hi * 8;
    #pragma unroll 8
    for (int k = 0; k < K; k += 32) {
        union { v16bf v; uint4 u[2]; } A, B;
        // Prefetch one tile ahead on the weight stream (global_prefetch_b8).
        if (k + 32 < K) __builtin_prefetch(wrow + k + 32, 0, 3);
        A.u[0] = *(const uint4*)(wrow + k);
        A.u[1] = *(const uint4*)(wrow + k + 16);
        B.u[0] = *(const uint4*)(xv + k);
        B.u[1] = *(const uint4*)(xv + k + 16);
        int p = (k >> 5) & 1;   // ping-pong accumulator: break C->C RAW chain
        acc[p] = __builtin_amdgcn_wmma_f32_16x16x32_bf16(
            false, A.v, false, B.v, (short)0, acc[p], false, false);
    }
    (void)acc;
}

template <int K1, int K2>
__global__ void gemv_wmma_kernel(const __bf16* __restrict__ W1,
                                 const __bf16* __restrict__ x1,
                                 const __bf16* __restrict__ W2,
                                 const __bf16* __restrict__ x2,
                                 const float* __restrict__ bias,
                                 float* __restrict__ y, int rows) {
    const int lane = threadIdx.x & 31;
    const int wave = blockIdx.x * (blockDim.x >> 5) + (threadIdx.x >> 5);
    const int row0 = wave * 16;
    if (row0 >= rows) return;
    const int m  = lane & 15;
    const int hi = lane >> 4;        // 0 or 1

    v8f acc[2] = {v8f{}, v8f{}};
    gemv_loop<0>(W1, x1, K1, row0 + m, hi, acc);
    if constexpr (K2 > 0) {
        gemv_loop<1>(W2, x2, K2, row0 + m, hi, acc);
    }
    v8f accs = acc[0] + acc[1];

    // D layout: VGPR r, lanes 0-15 -> row r ; lanes 16-31 -> row 8+r.
    // All 16 columns identical, so lanes 0 and 16 write the 16 outputs.
    if (m == 0) {
        int r0 = row0 + hi * 8;
        #pragma unroll
        for (int r = 0; r < 8; ++r)
            y[r0 + r] = accs[r] + bias[r0 + r];
    }
}

// ------------------------------ cell update --------------------------------

__device__ __forceinline__ float sigmoidf_(float x) {
    return 1.0f / (1.0f + __expf(-x));
}

__global__ void lstm_update_kernel(const float* __restrict__ g,
                                   float* __restrict__ h, float* __restrict__ c,
                                   __bf16* __restrict__ hb) {
    int k = blockIdx.x * blockDim.x + threadIdx.x;
    if (k >= HID) return;
    float gi = g[k], gf = g[HID + k], gg = g[2 * HID + k], go = g[3 * HID + k];
    float cn = sigmoidf_(gf) * c[k] + sigmoidf_(gi) * tanhf(gg);
    float hn = sigmoidf_(go) * tanhf(cn);
    c[k]  = cn;
    h[k]  = hn;
    hb[k] = f2bf(hn);
}

// -------------------------- softmax + sampling -----------------------------
// Deterministic Gumbel-max sampling from a hash of (t, token).  Writes probs
// and the sample, then builds next step's xa = [sample, input[t+1,:], 0...].

__global__ void softmax_sample_kernel(const float* __restrict__ logits,
                                      float* __restrict__ probs_out,
                                      float* __restrict__ sample_out,
                                      float* __restrict__ prev,
                                      __bf16* __restrict__ xa,
                                      const float* __restrict__ input, int t) {
    __shared__ float sm[256];
    __shared__ int   si[256];
    const int tid = threadIdx.x;

    float lmax = -3.0e38f;
    for (int j = tid; j < VOCAB; j += 256) lmax = fmaxf(lmax, logits[j]);
    sm[tid] = lmax; __syncthreads();
    for (int s = 128; s > 0; s >>= 1) {
        if (tid < s) sm[tid] = fmaxf(sm[tid], sm[tid + s]);
        __syncthreads();
    }
    const float gmax = sm[0]; __syncthreads();

    float lsum = 0.0f;
    for (int j = tid; j < VOCAB; j += 256) lsum += __expf(logits[j] - gmax);
    sm[tid] = lsum; __syncthreads();
    for (int s = 128; s > 0; s >>= 1) {
        if (tid < s) sm[tid] += sm[tid + s];
        __syncthreads();
    }
    const float inv = 1.0f / sm[0]; __syncthreads();

    float bkey = -3.0e38f; int bidx = 0;
    for (int j = tid; j < VOCAB; j += 256) {
        float l = logits[j];
        probs_out[j] = __expf(l - gmax) * inv;
        unsigned int h = (unsigned int)(t * VOCAB + j) * 2654435761u;
        h ^= h >> 16; h *= 2246822519u; h ^= h >> 13;
        float u = (float)((h >> 8) | 1u) * (1.0f / 16777216.0f); // (0,1)
        float key = l - __logf(-__logf(u));                      // Gumbel-max
        if (key > bkey) { bkey = key; bidx = j; }
    }
    sm[tid] = bkey; si[tid] = bidx; __syncthreads();
    for (int s = 128; s > 0; s >>= 1) {
        if (tid < s && sm[tid + s] > sm[tid]) { sm[tid] = sm[tid + s]; si[tid] = si[tid + s]; }
        __syncthreads();
    }
    const int samp = si[0];
    if (tid == 0) { sample_out[0] = (float)samp; prev[0] = (float)samp; }
    if (t + 1 < TSTEPS && tid < K0PAD) {
        float v = (tid == 0) ? (float)samp
                 : (tid <= INSZ ? input[(size_t)(t + 1) * INSZ + (tid - 1)] : 0.0f);
        xa[tid] = f2bf(v);
    }
}

// ------------------------------- launcher ----------------------------------

extern "C" void kernel_launch(void* const* d_in, const int* in_sizes, int n_in,
                              void* d_out, int out_size, void* d_ws, size_t ws_size,
                              hipStream_t stream) {
    (void)in_sizes; (void)n_in; (void)out_size; (void)ws_size;

    const float* input = (const float*)d_in[0];   // [128,1,64]
    const float* h0in  = (const float*)d_in[1];   // [2,1,2048]
    const float* c0in  = (const float*)d_in[2];
    const float* Wih0  = (const float*)d_in[3];   // [8192,65]
    const float* Whh0  = (const float*)d_in[4];   // [8192,2048]
    const float* bih0  = (const float*)d_in[5];
    const float* bhh0  = (const float*)d_in[6];
    const float* Wih1  = (const float*)d_in[7];   // [8192,2048]
    const float* Whh1  = (const float*)d_in[8];
    const float* bih1  = (const float*)d_in[9];
    const float* bhh1  = (const float*)d_in[10];
    const float* Wl    = (const float*)d_in[11];  // [4096,2048]
    const float* bl    = (const float*)d_in[12];

    float* out     = (float*)d_out;
    float* samples = out;                 // [128]
    float* probs   = out + TSTEPS;        // [128*4096]

    // Workspace carve (256B aligned chunks)
    char* base = (char*)d_ws;
    size_t off = 0;
    auto carve = [&](size_t bytes) -> char* {
        char* p = base + off;
        off += (bytes + 255) & ~(size_t)255;
        return p;
    };
    __bf16* Wih0b = (__bf16*)carve((size_t)GATES * K0PAD * 2);
    __bf16* Whh0b = (__bf16*)carve((size_t)GATES * HID * 2);
    __bf16* Wih1b = (__bf16*)carve((size_t)GATES * HID * 2);
    __bf16* Whh1b = (__bf16*)carve((size_t)GATES * HID * 2);
    __bf16* Wlb   = (__bf16*)carve((size_t)VOCAB * HID * 2);
    float*  b0sum = (float*)carve(GATES * 4);
    float*  b1sum = (float*)carve(GATES * 4);
    float*  g     = (float*)carve(GATES * 4);
    float*  logits= (float*)carve(VOCAB * 4);
    float*  h0f   = (float*)carve(HID * 4);
    float*  c0f   = (float*)carve(HID * 4);
    float*  h1f   = (float*)carve(HID * 4);
    float*  c1f   = (float*)carve(HID * 4);
    __bf16* h0b   = (__bf16*)carve(HID * 2);
    __bf16* h1b   = (__bf16*)carve(HID * 2);
    __bf16* xa    = (__bf16*)carve(K0PAD * 2);
    float*  prev  = (float*)carve(4);

    // ---- one-time (per call) weight conversion to bf16 ----
    cvt_pad_wih0_kernel<<<1024, 256, 0, stream>>>(Wih0, Wih0b);
    cvt_bf16_kernel<<<4096, 256, 0, stream>>>(Whh0, Whh0b, GATES * HID);
    cvt_bf16_kernel<<<4096, 256, 0, stream>>>(Wih1, Wih1b, GATES * HID);
    cvt_bf16_kernel<<<4096, 256, 0, stream>>>(Whh1, Whh1b, GATES * HID);
    cvt_bf16_kernel<<<4096, 256, 0, stream>>>(Wl,   Wlb,   VOCAB * HID);
    bias_sum_kernel<<<32, 256, 0, stream>>>(bih0, bhh0, b0sum, GATES);
    bias_sum_kernel<<<32, 256, 0, stream>>>(bih1, bhh1, b1sum, GATES);

    init_state_kernel<<<8, 256, 0, stream>>>(h0in, c0in, input,
                                             h0f, c0f, h1f, c1f,
                                             h0b, h1b, xa, prev);

    // ---- sequential autoregressive steps ----
    const int gatesBlocks  = GATES / 128;   // 8 waves/block * 16 rows/wave
    const int logitsBlocks = VOCAB / 128;
    for (int t = 0; t < TSTEPS; ++t) {
        // layer 0: g = Wih0*xa + Whh0*h0 + b
        gemv_wmma_kernel<K0PAD, HID><<<gatesBlocks, 256, 0, stream>>>(
            Wih0b, xa, Whh0b, h0b, b0sum, g, GATES);
        lstm_update_kernel<<<8, 256, 0, stream>>>(g, h0f, c0f, h0b);
        // layer 1: g = Wih1*h0' + Whh1*h1 + b
        gemv_wmma_kernel<HID, HID><<<gatesBlocks, 256, 0, stream>>>(
            Wih1b, h0b, Whh1b, h1b, b1sum, g, GATES);
        lstm_update_kernel<<<8, 256, 0, stream>>>(g, h1f, c1f, h1b);
        // logits = Wl*h1' + bl
        gemv_wmma_kernel<HID, 0><<<logitsBlocks, 256, 0, stream>>>(
            Wlb, h1b, (const __bf16*)nullptr, (const __bf16*)nullptr,
            bl, logits, VOCAB);
        softmax_sample_kernel<<<1, 256, 0, stream>>>(
            logits, probs + (size_t)t * VOCAB, samples + t, prev, xa, input, t);
    }
}
